// TransMIL_9637906612539
// MI455X (gfx1250) — matrix operand order repaired and checked by reading, not compiled, not run on hardware
//
#include <hip/hip_runtime.h>
#include <math.h>

typedef __attribute__((ext_vector_type(16))) _Float16 v16h;
typedef __attribute__((ext_vector_type(8)))  float    v8f;
typedef __attribute__((ext_vector_type(4)))  unsigned v4u;

#define NDIM   512
#define NTOK   2049
#define NROWS  8196          // 4 * 2049
#define TPAD   2080          // 65 * 32, padded key length
#define NCT    130           // TPAD / 16 column tiles
#define FMIN   (-3.4028235e38f)

// ---------------- helpers ----------------

union Frag { v16h v; unsigned u[8]; v4u q[2]; };

__device__ inline unsigned pack2h(float a, float b) {
    union { _Float16 h[2]; unsigned u; } r;
    r.h[0] = (_Float16)a; r.h[1] = (_Float16)b;
    return r.u;
}

__device__ inline v8f wmma16(const Frag& a, const Frag& b, v8f c) {
    return __builtin_amdgcn_wmma_f32_16x16x32_f16(
        false, a.v, false, b.v, (short)0, c, false, false);
}

__device__ inline float gelu_exact(float x) {
    return 0.5f * x * (1.0f + erff(x * 0.70710678118654752f));
}

// async copy 16 bytes global -> LDS (per lane), tracked by ASYNCcnt
__device__ inline void async_b128(unsigned lds_off, const _Float16* g) {
    asm volatile("global_load_async_to_lds_b128 %0, %1, off"
                 :: "v"(lds_off), "v"((unsigned long long)(size_t)g)
                 : "memory");
}
__device__ inline void wait_async0() {
    asm volatile("s_wait_asynccnt 0x0" ::: "memory");
}

// block of 256 threads (8 waves), sum-reduce
__device__ inline float block_sum_256(float v, float* red) {
#pragma unroll
    for (int o = 16; o > 0; o >>= 1) v += __shfl_xor(v, o, 32);
    int w = threadIdx.x >> 5;
    if ((threadIdx.x & 31) == 0) red[w] = v;
    __syncthreads();
    float t = (threadIdx.x < 8) ? red[threadIdx.x] : 0.0f;
#pragma unroll
    for (int o = 4; o > 0; o >>= 1) t += __shfl_xor(t, o, 32);
    if (threadIdx.x == 0) red[0] = t;
    __syncthreads();
    float r = red[0];
    __syncthreads();
    return r;
}

// ---------------- fp32 -> f16 converters ----------------

__global__ __launch_bounds__(256)
void cvt_kernel(const float* __restrict__ src, _Float16* __restrict__ dst, long n) {
    long i = ((long)blockIdx.x * 256 + threadIdx.x) * 8;
    if (i + 7 < n) {
        union { v4u q; unsigned u[4]; } o;
#pragma unroll
        for (int t = 0; t < 4; ++t) o.u[t] = pack2h(src[i + 2 * t], src[i + 2 * t + 1]);
        *(v4u*)(dst + i) = o.q;
    }
}

// W [K][N] fp32  ->  WT [N][K] f16
__global__ __launch_bounds__(256)
void cvtT_kernel(const float* __restrict__ W, _Float16* __restrict__ WT, int K, int N) {
    int n = blockIdx.x;
    for (int k = threadIdx.x; k < K; k += 256)
        WT[(long)n * K + k] = (_Float16)W[(long)k * N + n];
}

// ---------------- slide embedding table (25 x 512) ----------------

__global__ __launch_bounds__(256) void slide_emb_kernel(float* __restrict__ tab) {
    __shared__ float red[8];
    int t = blockIdx.x;
    float vals[2];
#pragma unroll
    for (int j = 0; j < 2; ++j) {
        int d = threadIdx.x + j * 256;
        double vec = pow(100.0, -2.0 * (double)(d >> 1) / 512.0);
        double arg = (double)t * vec;
        vals[j] = (d & 1) ? (float)cos(arg) : (float)sin(arg);
    }
    float mean = block_sum_256(vals[0] + vals[1], red) * (1.0f / 512.0f);
    float d0 = vals[0] - mean, d1 = vals[1] - mean;
    float stdv = sqrtf(block_sum_256(d0 * d0 + d1 * d1, red) / 511.0f);
    tab[t * NDIM + threadIdx.x]       = d0 / stdv;
    tab[t * NDIM + threadIdx.x + 256] = d1 / stdv;
}

// ---------------- WMMA GEMM: C = act(A[MxK] @ WT[NxK]^T + bias (+R)) --------
// A, WT are f16; mode bit0 = gelu, bit1 = +residual(f32), bit2 = f16 output.
// block = 256 thr = 8 waves; tile 64(M) x 128(N); K stepped by 32.
// Double-buffered LDS staged with async global->LDS b128 copies (ASYNCcnt):
// tile i+1 copies overlap tile i WMMA compute; one barrier per K-step.

#define ASTR 40
#define BSTR 40

__global__ __launch_bounds__(256)
void gemm_kernel(const _Float16* __restrict__ A, const _Float16* __restrict__ WT,
                 const float* __restrict__ bias, const float* __restrict__ R,
                 float* __restrict__ Cf, _Float16* __restrict__ Ch,
                 int M, int N, int K, int mode) {
    __shared__ __align__(16) _Float16 at[2][64 * ASTR];
    __shared__ __align__(16) _Float16 bt[2][128 * BSTR];

    int tid = threadIdx.x;
    int wave = tid >> 5, lane = tid & 31;
    int half = lane >> 4, l16 = lane & 15;
    int wr = wave >> 1, wc = wave & 1;           // 4 row-tiles x 2 col-halves
    int m0 = blockIdx.y * 64;
    int n0 = blockIdx.x * 128;

    // staging coordinates (one A chunk + two B chunks of 8 halves per thread)
    int sar = tid >> 2, sac = (tid & 3) * 8;     // A: row 0..63, col 0/8/16/24
    int gma = m0 + sar; if (gma >= M) gma = M - 1;
    const _Float16* agp = A + (long)gma * K + sac;
    int sb1n = (tid + 256) >> 2;
    const _Float16* bgp0 = WT + (long)(n0 + sar) * K + sac;
    const _Float16* bgp1 = WT + (long)(n0 + sb1n) * K + sac;
    unsigned aoff[2], boff0[2], boff1[2];
#pragma unroll
    for (int bufi = 0; bufi < 2; ++bufi) {
        aoff[bufi]  = (unsigned)(size_t)(void*)&at[bufi][sar * ASTR + sac];
        boff0[bufi] = (unsigned)(size_t)(void*)&bt[bufi][sar * BSTR + sac];
        boff1[bufi] = (unsigned)(size_t)(void*)&bt[bufi][sb1n * BSTR + sac];
    }

    v8f zero = {0.f,0.f,0.f,0.f,0.f,0.f,0.f,0.f};
    v8f acc[4] = {zero, zero, zero, zero};

    int nt = K >> 5;
    // prologue: issue tile 0 into buffer 0
    async_b128(aoff[0],  agp);
    async_b128(boff0[0], bgp0);
    async_b128(boff1[0], bgp1);
    wait_async0();
    __syncthreads();

    for (int i = 0; i < nt; ++i) {
        int cur = i & 1;
        if (i + 1 < nt) {                       // prefetch next tile (overlaps WMMA)
            int kk = (i + 1) << 5;
            async_b128(aoff[cur ^ 1],  agp  + kk);
            async_b128(boff0[cur ^ 1], bgp0 + kk);
            async_b128(boff1[cur ^ 1], bgp1 + kk);
        }
        Frag af;
        {
            const _Float16* ar = &at[cur][(wr * 16 + l16) * ASTR + half * 8];
            af.q[0] = *(const v4u*)ar;          // elems 0..7  : k = half*8 + 0..7
            af.q[1] = *(const v4u*)(ar + 16);   // elems 8..15 : k = 16 + half*8 + 0..7
        }
#pragma unroll
        for (int c = 0; c < 4; ++c) {
            Frag bf;
            const _Float16* br = &bt[cur][(wc * 64 + c * 16 + l16) * BSTR + half * 8];
            bf.q[0] = *(const v4u*)br;          // k = half*8 + 0..7
            bf.q[1] = *(const v4u*)(br + 16);   // k = 16 + half*8 + 0..7
            acc[c] = wmma16(af, bf, acc[c]);
        }
        if (i + 1 < nt) {
            wait_async0();                      // next tile landed
            __syncthreads();                    // + everyone done reading cur
        }
    }

#pragma unroll
    for (int c = 0; c < 4; ++c) {
#pragma unroll
        for (int r = 0; r < 8; ++r) {
            int m = m0 + wr * 16 + r + half * 8;
            int n = n0 + wc * 64 + c * 16 + l16;
            if (m < M) {
                float v = acc[c][r];
                if (bias) v += bias[n];
                if (mode & 1) v = gelu_exact(v);
                if (mode & 2) v += R[(long)m * N + n];
                if (mode & 4) Ch[(long)m * N + n] = (_Float16)v;
                else          Cf[(long)m * N + n] = v;
            }
        }
    }
}

// ---------------- layernorm over rows of 512 (f32 + f16 outputs) ------------

__global__ __launch_bounds__(256)
void ln_kernel(const float* __restrict__ x, const float* __restrict__ g,
               const float* __restrict__ bb, float* __restrict__ y,
               _Float16* __restrict__ yh) {
    __shared__ float red[8];
    long row = blockIdx.x;
    const float* xr = x + row * NDIM;
    float a = xr[threadIdx.x], b = xr[threadIdx.x + 256];
    float mean = block_sum_256(a + b, red) * (1.0f / 512.0f);
    float da = a - mean, db = b - mean;
    float var = block_sum_256(da * da + db * db, red) * (1.0f / 512.0f);
    float r = rsqrtf(var + 1e-5f);
    float o0 = da * r * g[threadIdx.x]       + bb[threadIdx.x];
    float o1 = db * r * g[threadIdx.x + 256] + bb[threadIdx.x + 256];
    y[row * NDIM + threadIdx.x]        = o0;
    y[row * NDIM + threadIdx.x + 256]  = o1;
    yh[row * NDIM + threadIdx.x]       = (_Float16)o0;
    yh[row * NDIM + threadIdx.x + 256] = (_Float16)o1;
}

// ---------------- embedding add (bilinear pos + slide emb) ----------------

__global__ __launch_bounds__(256)
void embed_kernel(const float* __restrict__ fcout, const float* __restrict__ coords,
                  const int* __restrict__ slide_ids, const float* __restrict__ pos_emb,
                  const float* __restrict__ slide_emb, float* __restrict__ xbuf) {
    int tok = blockIdx.x;                 // 0..8191
    int b = tok >> 11, n = tok & 2047;
    float cx = coords[tok * 2 + 0];       // * (DENS-1) == * 1
    float cy = coords[tok * 2 + 1];
    int fx = (int)floorf(cx), fy = (int)floorf(cy);
    int x0 = min(max(fx, 0), 1),     x1 = min(max(fx + 1, 0), 1);
    int y0 = min(max(fy, 0), 1),     y1 = min(max(fy + 1, 0), 1);
    float frx = cx - (float)x0, fry = cy - (float)y0;
    float w00 = (1.f - frx) * (1.f - fry), w01 = (1.f - frx) * fry;
    float w10 = frx * (1.f - fry),         w11 = frx * fry;
    int sid = slide_ids[tok];
    long orow = (long)(b * NTOK + 1 + n) * NDIM;
    for (int d = threadIdx.x; d < NDIM; d += 256) {
        float p = w00 * pos_emb[(x0 * 2 + y0) * NDIM + d]
                + w01 * pos_emb[(x0 * 2 + y1) * NDIM + d]
                + w10 * pos_emb[(x1 * 2 + y0) * NDIM + d]
                + w11 * pos_emb[(x1 * 2 + y1) * NDIM + d];
        xbuf[orow + d] = fcout[(long)tok * NDIM + d] + p + slide_emb[sid * NDIM + d];
    }
}

__global__ void cls_kernel(const float* __restrict__ cls, float* __restrict__ xbuf) {
    int b = blockIdx.x;
    for (int d = threadIdx.x; d < NDIM; d += blockDim.x)
        xbuf[(long)b * NTOK * NDIM + d] = cls[d];
}

// ---------------- attention: one block per (query-16-tile, head, batch) -----
// qkv is f16. dyn LDS: scores 16 x TPAD fp32 + V^T tile 64x40 f16.

#define VSTR 40
#define ATTN_SMEM (16 * TPAD * 4 + 64 * VSTR * 2)

__global__ __launch_bounds__(256)
void attn_kernel(const _Float16* __restrict__ qkv, _Float16* __restrict__ obuf,
                 const int* __restrict__ lens) {
    extern __shared__ float smem[];
    float* sc = smem;                                   // [16][TPAD]
    _Float16* vt = (_Float16*)(sc + 16 * TPAD);         // [64][VSTR]  (n=d, k=token)

    int qt = blockIdx.x, h = blockIdx.y, b = blockIdx.z;
    int q0 = qt * 16;
    int Lb = lens[b] + 1;
    int tid = threadIdx.x, wave = tid >> 5, lane = tid & 31;
    int half = lane >> 4, l16 = lane & 15;
    long base = (long)b * NTOK * 1536;

    // ---- Q fragments (k=0..31 and k=32..63), per-wave copy ----
    Frag qa0, qa1;
    {
        int tok = q0 + l16; if (tok > NTOK - 1) tok = NTOK - 1;
        const _Float16* qr = qkv + base + (long)tok * 1536 + h * 64;
        qa0.q[0] = *(const v4u*)(qr + half * 8);
        qa0.q[1] = *(const v4u*)(qr + 16 + half * 8);
        qa1.q[0] = *(const v4u*)(qr + 32 + half * 8);
        qa1.q[1] = *(const v4u*)(qr + 48 + half * 8);
    }

    // ---- phase 1: S = Q K^T * scale, masked, into LDS ----
    for (int j = wave; j < NCT; j += 8) {
        Frag kb0, kb1;
        int ntok = j * 16 + l16;
        int tokc = (ntok > NTOK - 1) ? 0 : ntok;          // clamped; masked below
        const _Float16* kr = qkv + base + (long)tokc * 1536 + 512 + h * 64;
        kb0.q[0] = *(const v4u*)(kr + half * 8);
        kb0.q[1] = *(const v4u*)(kr + half * 8 + 16);
        kb1.q[0] = *(const v4u*)(kr + 32 + half * 8);
        kb1.q[1] = *(const v4u*)(kr + 32 + half * 8 + 16);
        v8f acc = {0.f,0.f,0.f,0.f,0.f,0.f,0.f,0.f};
        acc = wmma16(qa0, kb0, acc);
        acc = wmma16(qa1, kb1, acc);
#pragma unroll
        for (int r = 0; r < 8; ++r) {
            int m = r + half * 8;
            int qtok = q0 + m;
            int ktok = j * 16 + l16;
            float s = acc[r] * 0.125f;                    // DH^-0.5
            bool ok = (qtok < Lb) && (ktok < Lb);
            sc[m * TPAD + j * 16 + l16] = ok ? s : FMIN;
        }
    }
    __syncthreads();

    // ---- phase 2: softmax along each of 16 rows (16 threads / row) ----
    {
        int row = tid >> 4, c0 = tid & 15;
        float* srow = sc + row * TPAD;
        float mx = FMIN;
        for (int c = c0; c < TPAD; c += 16) mx = fmaxf(mx, srow[c]);
#pragma unroll
        for (int o = 8; o > 0; o >>= 1) mx = fmaxf(mx, __shfl_xor(mx, o, 32));
        float sum = 0.f;
        for (int c = c0; c < TPAD; c += 16) {
            float e = expf(srow[c] - mx);
            srow[c] = e; sum += e;
        }
#pragma unroll
        for (int o = 8; o > 0; o >>= 1) sum += __shfl_xor(sum, o, 32);
        float is = 1.0f / sum;
        for (int c = c0; c < TPAD; c += 16) srow[c] *= is;
    }
    __syncthreads();

    // ---- phase 3: O = P @ V ; waves 0..3 each own a 16-wide d-tile ----
    v8f oacc = {0.f,0.f,0.f,0.f,0.f,0.f,0.f,0.f};
    int dt = wave;
    int skk = tid >> 3, sdc = (tid & 7) * 8;   // V staging: token-local, d-chunk
    for (int kt = 0; kt < TPAD / 32; ++kt) {
        // stage V^T tile: vt[d][k] (transpose in registers, 8 halves per thread)
        {
            int tok = kt * 32 + skk;
            union { v4u q; _Float16 hh[8]; } u8;
            u8.q = (v4u){0u, 0u, 0u, 0u};
            if (tok <= NTOK - 1)
                u8.q = *(const v4u*)(qkv + base + (long)tok * 1536 + 1024 + h * 64 + sdc);
#pragma unroll
            for (int e = 0; e < 8; ++e) vt[(sdc + e) * VSTR + skk] = u8.hh[e];
        }
        __syncthreads();
        if (wave < 4) {
            Frag pa, vb;
            const float* srow = sc + l16 * TPAD + kt * 32;
#pragma unroll
            for (int t = 0; t < 4; ++t) {
                int k = half * 8 + 2 * t;
                pa.u[t]     = pack2h(srow[k],      srow[k + 1]);
                pa.u[4 + t] = pack2h(srow[16 + k], srow[17 + k]);
            }
            const _Float16* vr = &vt[(dt * 16 + l16) * VSTR + half * 8];
            vb.q[0] = *(const v4u*)vr;
            vb.q[1] = *(const v4u*)(vr + 16);
            oacc = wmma16(pa, vb, oacc);
        }
        __syncthreads();
    }
    if (wave < 4) {
#pragma unroll
        for (int r = 0; r < 8; ++r) {
            int m = r + half * 8;
            int tok = q0 + m;
            if (tok <= NTOK - 1)
                obuf[((long)b * NTOK + tok) * NDIM + h * 64 + dt * 16 + l16] =
                    (_Float16)oacc[r];
        }
    }
}

// ---------------- classifier head ----------------

__global__ __launch_bounds__(256)
void head_kernel(const float* __restrict__ xn, const float* __restrict__ hw,
                 const float* __restrict__ hb, float* __restrict__ out) {
    __shared__ float red[8];
    int b = blockIdx.x >> 1, c = blockIdx.x & 1;
    const float* xr = xn + (long)b * NTOK * NDIM;       // cls row
    float s = 0.f;
    for (int d = threadIdx.x; d < NDIM; d += 256) s += xr[d] * hw[d * 2 + c];
    s = block_sum_256(s, red);
    if (threadIdx.x == 0) out[b * 2 + c] = s + hb[c];
}

// ---------------- orchestration ----------------

static inline size_t alignup(size_t v) { return (v + 255) & ~(size_t)255; }

extern "C" void kernel_launch(void* const* d_in, const int* in_sizes, int n_in,
                              void* d_out, int out_size, void* d_ws, size_t ws_size,
                              hipStream_t stream) {
    const float* x         = (const float*)d_in[0];
    const float* coords    = (const float*)d_in[1];
    const int*   lens      = (const int*)  d_in[2];
    const int*   slide_ids = (const int*)  d_in[3];
    const float* fc_w      = (const float*)d_in[4];
    const float* fc_b      = (const float*)d_in[5];
    const float* cls_token = (const float*)d_in[6];
    const float* pos_emb   = (const float*)d_in[7];
    const float* ln1_g     = (const float*)d_in[8];
    const float* ln1_b     = (const float*)d_in[9];
    const float* wqkv      = (const float*)d_in[10];
    const float* wo_w      = (const float*)d_in[11];
    const float* wo_b      = (const float*)d_in[12];
    const float* ln2_g     = (const float*)d_in[13];
    const float* ln2_b     = (const float*)d_in[14];
    const float* w1        = (const float*)d_in[15];
    const float* b1        = (const float*)d_in[16];
    const float* w2        = (const float*)d_in[17];
    const float* b2        = (const float*)d_in[18];
    const float* lnf_g     = (const float*)d_in[19];
    const float* lnf_b     = (const float*)d_in[20];
    const float* head_w    = (const float*)d_in[21];
    const float* head_b    = (const float*)d_in[22];
    float* out = (float*)d_out;
    (void)ws_size; (void)n_in; (void)in_sizes; (void)out_size;

    char* wsb = (char*)d_ws;
    size_t o = 0;
    float* slide = (float*)(wsb + o); o = alignup(o + 25 * NDIM * 4);
    float* tmp   = (float*)(wsb + o); o = alignup(o + (size_t)8192 * NDIM * 4);
    float* xbuf  = (float*)(wsb + o); o = alignup(o + (size_t)NROWS * NDIM * 4);
    float* xn    = (float*)(wsb + o); o = alignup(o + (size_t)NROWS * NDIM * 4);
    _Float16* xh     = (_Float16*)(wsb + o); o = alignup(o + (size_t)8192 * 768 * 2);
    _Float16* fcwT   = (_Float16*)(wsb + o); o = alignup(o + (size_t)512 * 768 * 2);
    _Float16* wqkvT  = (_Float16*)(wsb + o); o = alignup(o + (size_t)2 * 1536 * 512 * 2);
    _Float16* wowT   = (_Float16*)(wsb + o); o = alignup(o + (size_t)2 * 512 * 512 * 2);
    _Float16* w1T    = (_Float16*)(wsb + o); o = alignup(o + (size_t)2 * 2048 * 512 * 2);
    _Float16* w2T    = (_Float16*)(wsb + o); o = alignup(o + (size_t)2 * 512 * 2048 * 2);
    _Float16* xnh    = (_Float16*)(wsb + o); o = alignup(o + (size_t)NROWS * NDIM * 2);
    _Float16* qkvh   = (_Float16*)(wsb + o); o = alignup(o + (size_t)NROWS * 1536 * 2);
    _Float16* obufh  = (_Float16*)(wsb + o); o = alignup(o + (size_t)NROWS * NDIM * 2);
    _Float16* hbufh  = (_Float16*)(wsb + o); o = alignup(o + (size_t)NROWS * 2048 * 2);

    hipFuncSetAttribute(reinterpret_cast<const void*>(attn_kernel),
                        hipFuncAttributeMaxDynamicSharedMemorySize, ATTN_SMEM);

    // constant tables / operand conversion
    slide_emb_kernel<<<25, 256, 0, stream>>>(slide);
    {
        long nx = (long)8192 * 768;
        cvt_kernel<<<(unsigned)((nx / 8 + 255) / 256), 256, 0, stream>>>(x, xh, nx);
    }
    cvtT_kernel<<<512, 256, 0, stream>>>(fc_w, fcwT, 768, 512);
    for (int i = 0; i < 2; ++i) {
        cvtT_kernel<<<1536, 256, 0, stream>>>(wqkv + (size_t)i * 512 * 1536,
                                              wqkvT + (size_t)i * 1536 * 512, 512, 1536);
        cvtT_kernel<<<512, 256, 0, stream>>>(wo_w + (size_t)i * 512 * 512,
                                             wowT + (size_t)i * 512 * 512, 512, 512);
        cvtT_kernel<<<2048, 256, 0, stream>>>(w1 + (size_t)i * 512 * 2048,
                                              w1T + (size_t)i * 2048 * 512, 512, 2048);
        cvtT_kernel<<<512, 256, 0, stream>>>(w2 + (size_t)i * 2048 * 512,
                                             w2T + (size_t)i * 512 * 2048, 2048, 512);
    }

    // fc + gelu : [8192,768] @ [768,512] -> tmp (f32)
    gemm_kernel<<<dim3(512 / 128, 8192 / 64), 256, 0, stream>>>(
        xh, fcwT, fc_b, nullptr, tmp, nullptr, 8192, 512, 768, 1);

    embed_kernel<<<8192, 256, 0, stream>>>(tmp, coords, slide_ids, pos_emb, slide, xbuf);
    cls_kernel<<<4, 256, 0, stream>>>(cls_token, xbuf);

    int mt = (NROWS + 63) / 64;     // 129
    for (int i = 0; i < 2; ++i) {
        // attention block
        ln_kernel<<<NROWS, 256, 0, stream>>>(xbuf, ln1_g + i * NDIM, ln1_b + i * NDIM,
                                             xn, xnh);
        gemm_kernel<<<dim3(1536 / 128, mt), 256, 0, stream>>>(
            xnh, wqkvT + (size_t)i * 1536 * 512, nullptr, nullptr,
            nullptr, qkvh, NROWS, 1536, 512, 4);
        attn_kernel<<<dim3((NTOK + 15) / 16, 8, 4), 256, ATTN_SMEM, stream>>>(
            qkvh, obufh, lens);
        gemm_kernel<<<dim3(512 / 128, mt), 256, 0, stream>>>(
            obufh, wowT + (size_t)i * 512 * 512, wo_b + i * NDIM, xbuf,
            xbuf, nullptr, NROWS, 512, 512, 2);
        // mlp block
        ln_kernel<<<NROWS, 256, 0, stream>>>(xbuf, ln2_g + i * NDIM, ln2_b + i * NDIM,
                                             xn, xnh);
        gemm_kernel<<<dim3(2048 / 128, mt), 256, 0, stream>>>(
            xnh, w1T + (size_t)i * 2048 * 512, b1 + i * 2048, nullptr,
            nullptr, hbufh, NROWS, 2048, 512, 5);
        gemm_kernel<<<dim3(512 / 128, mt), 256, 0, stream>>>(
            hbufh, w2T + (size_t)i * 512 * 2048, b2 + i * NDIM, xbuf,
            xbuf, nullptr, NROWS, 512, 2048, 2);
    }

    // final LN + cls head
    ln_kernel<<<NROWS, 256, 0, stream>>>(xbuf, lnf_g, lnf_b, xn, xnh);
    head_kernel<<<8, 256, 0, stream>>>(xn, head_w, head_b, out);
}
